// Transformer_66915590472143
// MI455X (gfx1250) — matrix operand order repaired and checked by reading, not compiled, not run on hardware
//
#include <hip/hip_runtime.h>
#include <math.h>
#include <stdint.h>

// ---------------------------------------------------------------------------
// Transformer encoder (B=32,S=600,F=9,DV=1024,H=16,DH=64,L=2) for gfx1250.
// GEMMs + attention on v_wmma_f32_16x16x32_f16; tile fills use CDNA5
// GLOBAL_LOAD_ASYNC_TO_LDS_B128 with double-buffered LDS (ASYNCcnt).
// ---------------------------------------------------------------------------

typedef __attribute__((ext_vector_type(16))) _Float16 v16h;
typedef __attribute__((ext_vector_type(8)))  _Float16 v8h;
typedef __attribute__((ext_vector_type(8)))  float    v8f;

#define B_   32
#define S_   600
#define F_   9
#define DV_  1024
#define H_   16
#define DH_  64
#define L_   2
#define BS_  (B_ * S_)          // 19200 tokens

// CDNA5 async global->LDS copy (16B per lane), tracked by ASYNCcnt.
// LDS operand = low 32 bits of the generic shared-memory pointer
#define ASYNC_B128(gp, lp)                                                   \
  asm volatile("global_load_async_to_lds_b128 %0, %1, off"                   \
               :: "v"((unsigned)(uintptr_t)(lp)), "v"(gp) : "memory")
#define WAIT_ASYNC() asm volatile("s_wait_asynccnt 0x0" ::: "memory")

static __device__ __forceinline__ v8h zero8h() {
  v8h z;
#pragma unroll
  for (int i = 0; i < 8; i++) z[i] = (_Float16)0;
  return z;
}
static __device__ __forceinline__ v8f zero8f() {
  v8f z;
#pragma unroll
  for (int i = 0; i < 8; i++) z[i] = 0.0f;
  return z;
}
// Build a WMMA v16h fragment from two aligned 16-byte LDS chunks.
static __device__ __forceinline__ v16h frag16(const _Float16* p, int off2_halves) {
  v8h lo = *(const v8h*)p;
  v8h hi = *(const v8h*)(p + off2_halves);
  return __builtin_shufflevector(lo, hi, 0,1,2,3,4,5,6,7,8,9,10,11,12,13,14,15);
}

// ---------------------------------------------------------------------------
// Small helper kernels (conversion / tiny projections / layernorm)
// ---------------------------------------------------------------------------

__global__ void k_enc_in(const float* __restrict__ x, const float* __restrict__ W,
                         const float* __restrict__ b, float* __restrict__ h32) {
  long long gid = (long long)blockIdx.x * 256 + threadIdx.x;   // BS_*DV_ threads
  int t = (int)(gid >> 10);
  int c = (int)(gid & 1023);
  float acc = b[c];
#pragma unroll
  for (int f = 0; f < F_; f++) acc += x[t * F_ + f] * W[f * DV_ + c];
  h32[gid] = acc;
}

__global__ void k_out_proj(const float* __restrict__ h32, const float* __restrict__ W,
                           const float* __restrict__ b, float* __restrict__ out) {
  int gid = blockIdx.x * 256 + threadIdx.x;
  if (gid >= BS_ * F_) return;
  int t = gid / F_;
  int f = gid - t * F_;
  const float* hr = h32 + (size_t)t * DV_;
  float acc = b[f];
  for (int c = 0; c < DV_; c++) acc += hr[c] * W[c * F_ + f];
  out[gid] = acc;
}

__global__ void k_f32_to_f16(const float* __restrict__ in, _Float16* __restrict__ out,
                             int n) {
  int gid = blockIdx.x * 256 + threadIdx.x;
  if (gid < n) out[gid] = (_Float16)in[gid];
}

// Weight [K=DV][N=DV] row-major -> f16 transposed Bt[n][k]
__global__ void k_w_to_f16_T(const float* __restrict__ in, _Float16* __restrict__ out) {
  int gid = blockIdx.x * 256 + threadIdx.x;       // DV_*DV_ threads
  int n = gid >> 10;
  int k = gid & 1023;
  out[(size_t)n * DV_ + k] = (_Float16)in[(size_t)k * DV_ + n];
}

// Wq[l] layout [H][DV][DH]; logical matrix W[k=dv][n=h*DH+d]; emit Bt[n][k] f16.
__global__ void k_qkv_w_to_f16_T(const float* __restrict__ in, _Float16* __restrict__ out) {
  int gid = blockIdx.x * 256 + threadIdx.x;       // DV_*DV_ threads
  int n  = gid >> 10;                             // h*64 + d
  int dv = gid & 1023;
  int hq = n >> 6;
  int d  = n & 63;
  out[gid] = (_Float16)in[((size_t)hq * DV_ + dv) * DH_ + d];
}

// out = LayerNorm(a + b) * g + beta ; also write f16 copy. One block per token.
__global__ __launch_bounds__(256)
void k_add_ln(const float* __restrict__ a, const float* __restrict__ b,
              const float* __restrict__ g, const float* __restrict__ be,
              float* __restrict__ out32, _Float16* __restrict__ out16) {
  __shared__ float r1[256], r2[256];
  int t = blockIdx.x;
  const float* pa = a + (size_t)t * DV_;
  const float* pb = b + (size_t)t * DV_;
  float loc[4], s = 0.f, s2 = 0.f;
#pragma unroll
  for (int i = 0; i < 4; i++) {
    int c = threadIdx.x + i * 256;
    float v = pa[c] + pb[c];
    loc[i] = v; s += v; s2 += v * v;
  }
  r1[threadIdx.x] = s; r2[threadIdx.x] = s2;
  __syncthreads();
  for (int st = 128; st > 0; st >>= 1) {
    if (threadIdx.x < st) { r1[threadIdx.x] += r1[threadIdx.x + st];
                            r2[threadIdx.x] += r2[threadIdx.x + st]; }
    __syncthreads();
  }
  float mu  = r1[0] * (1.0f / DV_);
  float var = r2[0] * (1.0f / DV_) - mu * mu;
  float ri  = rsqrtf(var + 1e-5f);
#pragma unroll
  for (int i = 0; i < 4; i++) {
    int c = threadIdx.x + i * 256;
    float v = (loc[i] - mu) * ri * g[c] + be[c];
    out32[(size_t)t * DV_ + c] = v;
    out16[(size_t)t * DV_ + c] = (_Float16)v;
  }
}

// ---------------------------------------------------------------------------
// WMMA GEMM: C[M,N] = A[M,K](f16,row-major) x Bt[N,K](f16,B-transposed) + bias
// 256 threads (8 waves), 128x128 tile, K-tile 64 (2x WMMA substeps of 32),
// wave tile 64x32. Double-buffered LDS, async tile fills (ASYNCcnt), one
// barrier per K-tile. REQUIRES M%128==0, N%128==0, K%64==0 (true here).
// MODE 0: f32 row-major; MODE 1: f16 QKV [B,H,S,DH]; MODE 2: f16 + ELU.
// ---------------------------------------------------------------------------
template <int MODE>
__global__ __launch_bounds__(256)
void k_gemm(const _Float16* __restrict__ A, const _Float16* __restrict__ Bt,
            const float* __restrict__ bias, void* __restrict__ outp,
            int M, int N, int K) {
  constexpr int BM = 128, BN = 128, BK = 64, LD = 72;   // LD padded (bank-free)
  __shared__ __align__(16) _Float16 As[2][BM * LD];
  __shared__ __align__(16) _Float16 Bs[2][BN * LD];

  const int tid  = threadIdx.x;
  const int lane = tid & 31;
  const int wid  = tid >> 5;
  const int wm   = wid >> 2;          // 0..1  (64 rows each)
  const int wn   = wid & 3;           // 0..3  (32 cols each)
  const int lrow = lane & 15;
  const int lhi  = lane >> 4;         // 0/1 half-wave
  const int bm0  = blockIdx.y * BM;
  const int bn0  = blockIdx.x * BN;

  // Per-thread async-copy assignment: 4 chunks of A + 4 of B per K-tile.
  const int r0 = tid >> 3;            // 0..31, rows r0 + i*32
  const int ch = (tid & 7) * 8;       // half offset within the 64-wide K slice

  auto issue_tile = [&](int buf, int kt) {
#pragma unroll
    for (int i = 0; i < 4; i++) {
      int row = r0 + i * 32;
      ASYNC_B128(A  + (size_t)(bm0 + row) * K + kt + ch, &As[buf][row * LD + ch]);
      ASYNC_B128(Bt + (size_t)(bn0 + row) * K + kt + ch, &Bs[buf][row * LD + ch]);
    }
  };

  v8f acc[4][2];
#pragma unroll
  for (int i = 0; i < 4; i++)
#pragma unroll
    for (int j = 0; j < 2; j++) acc[i][j] = zero8f();

  issue_tile(0, 0);
  WAIT_ASYNC();
  __syncthreads();

  const int NKT = K / BK;
  for (int t = 0; t < NKT; t++) {
    const int cur = t & 1;
    if (t + 1 < NKT) issue_tile(cur ^ 1, (t + 1) * BK);   // prefetch next tile

#pragma unroll
    for (int ks = 0; ks < 2; ks++) {                      // 2 x 32-wide substeps
      v16h af[4], bf[2];
#pragma unroll
      for (int mf = 0; mf < 4; mf++) {
        const _Float16* p =
            &As[cur][(wm * 64 + mf * 16 + lrow) * LD + ks * 32 + lhi * 8];
        af[mf] = frag16(p, 16);
      }
#pragma unroll
      for (int nf = 0; nf < 2; nf++) {
        const _Float16* p =
            &Bs[cur][(wn * 32 + nf * 16 + lrow) * LD + ks * 32 + lhi * 8];
        bf[nf] = frag16(p, 16);
      }
#pragma unroll
      for (int mf = 0; mf < 4; mf++)
#pragma unroll
        for (int nf = 0; nf < 2; nf++)
          acc[mf][nf] = __builtin_amdgcn_wmma_f32_16x16x32_f16(
              false, af[mf], false, bf[nf], (short)0, acc[mf][nf], false, false);
    }

    WAIT_ASYNC();        // next tile landed (this wave's copies)
    __syncthreads();     // all waves done copying & consuming
  }

#pragma unroll
  for (int mf = 0; mf < 4; mf++)
#pragma unroll
    for (int nf = 0; nf < 2; nf++)
#pragma unroll
      for (int r = 0; r < 8; r++) {
        int row = bm0 + wm * 64 + mf * 16 + r + (lhi << 3);
        int col = bn0 + wn * 32 + nf * 16 + lrow;
        float v = acc[mf][nf][r] + bias[col];
        if constexpr (MODE == 0) {
          ((float*)outp)[(size_t)row * N + col] = v;
        } else if constexpr (MODE == 1) {
          int bb = row / S_;
          int ss = row - bb * S_;
          int hq = col >> 6;
          int dd = col & 63;
          ((_Float16*)outp)[(((size_t)(bb * H_ + hq)) * S_ + ss) * DH_ + dd] = (_Float16)v;
        } else {  // MODE == 2: ELU -> f16
          float e = v > 0.f ? v : (__expf(v) - 1.f);
          ((_Float16*)outp)[(size_t)row * N + col] = (_Float16)e;
        }
      }
}

// ---------------------------------------------------------------------------
// Flash attention. Grid: (ceil(S/64), H, B). 128 threads = 4 waves.
// Q/K/V f16 in [B,H,S,DH]; ctx written f16 head-interleaved: col = d*H + h.
// K tile filled with async-to-LDS copies (stale ragged-edge rows are masked
// to -inf before softmax, so no zero-fill is needed on that path).
// Fragments are preloaded in batches of 4 so DScnt waits amortize.
// ---------------------------------------------------------------------------
__global__ __launch_bounds__(128)
void k_attn(const _Float16* __restrict__ Q, const _Float16* __restrict__ Km,
            const _Float16* __restrict__ V, _Float16* __restrict__ ctx) {
  constexpr int LD = 72;                       // padded stride (halves)
  __shared__ __align__(16) _Float16 Qs[64 * LD];
  __shared__ __align__(16) _Float16 Ks[64 * LD];
  __shared__ __align__(16) _Float16 Vs[64 * LD];   // transposed: Vs[d][key]
  __shared__ __align__(16) _Float16 Ps[64 * LD];

  const int qb = blockIdx.x, hh = blockIdx.y, bb = blockIdx.z;
  const size_t base = ((size_t)(bb * H_ + hh)) * S_ * DH_;
  const int tid  = threadIdx.x;
  const int lane = tid & 31;
  const int wv   = tid >> 5;
  const int lrow = lane & 15;
  const int lhi  = lane >> 4;

  // Load Q block (64 rows x 64) once (zero-filled at the ragged edge).
#pragma unroll
  for (int i = 0; i < 4; i++) {
    int c = tid + i * 128;          // 512 chunks
    int row = c >> 3, chq = c & 7;
    int sg = qb * 64 + row;
    v8h v = zero8h();
    if (sg < S_) v = *(const v8h*)(Q + base + (size_t)sg * DH_ + chq * 8);
    *(v8h*)&Qs[row * LD + chq * 8] = v;
  }
  __syncthreads();
  v16h aq[2];
#pragma unroll
  for (int ks = 0; ks < 2; ks++)
    aq[ks] = frag16(&Qs[(wv * 16 + lrow) * LD + ks * 32 + lhi * 8], 16);

  float mi[8], li[8];
#pragma unroll
  for (int r = 0; r < 8; r++) { mi[r] = -3.0e38f; li[r] = 0.f; }
  v8f o[4];
#pragma unroll
  for (int nn = 0; nn < 4; nn++) o[nn] = zero8f();

  const float scale = 0.125f;      // 1/sqrt(DH)

  for (int j = 0; j < (S_ + 63) / 64; j++) {
    __syncthreads();               // protect Ks/Vs reuse
#pragma unroll
    for (int i = 0; i < 4; i++) {
      int c = tid + i * 128;
      int row = c >> 3, chq = c & 7;
      int kg = j * 64 + row;
      if (kg < S_)
        ASYNC_B128(Km + base + (size_t)kg * DH_ + chq * 8, &Ks[row * LD + chq * 8]);
      v8h vv = zero8h();
      if (kg < S_) vv = *(const v8h*)(V + base + (size_t)kg * DH_ + chq * 8);
#pragma unroll
      for (int e = 0; e < 8; e++) Vs[(chq * 8 + e) * LD + row] = vv[e];   // transpose
    }
    WAIT_ASYNC();
    __syncthreads();

    // S = Q * K^T  (per wave: 16 q-rows x 64 keys); batch-preload B frags.
    v8f sA[4];
#pragma unroll
    for (int nf = 0; nf < 4; nf++) sA[nf] = zero8f();
#pragma unroll
    for (int ks = 0; ks < 2; ks++) {
      v16h bk[4];
#pragma unroll
      for (int nf = 0; nf < 4; nf++)
        bk[nf] = frag16(&Ks[(nf * 16 + lrow) * LD + ks * 32 + lhi * 8], 16);
#pragma unroll
      for (int nf = 0; nf < 4; nf++)
        sA[nf] = __builtin_amdgcn_wmma_f32_16x16x32_f16(
            false, aq[ks], false, bk[nf], (short)0, sA[nf], false, false);
    }

    // Online softmax update (rows map to VGPR index r within each half-wave).
#pragma unroll
    for (int r = 0; r < 8; r++) {
      float rowm = -3.0e38f;
#pragma unroll
      for (int nf = 0; nf < 4; nf++) {
        int cg = j * 64 + nf * 16 + lrow;
        float xx = (cg < S_) ? sA[nf][r] * scale : -3.0e38f;
        sA[nf][r] = xx;
        rowm = fmaxf(rowm, xx);
      }
#pragma unroll
      for (int off = 1; off < 16; off <<= 1)
        rowm = fmaxf(rowm, __shfl_xor(rowm, off, 32));
      float mn = fmaxf(mi[r], rowm);
      float al = __expf(mi[r] - mn);
      mi[r] = mn;
      float rs = 0.f;
#pragma unroll
      for (int nf = 0; nf < 4; nf++) {
        float pv = __expf(sA[nf][r] - mn);
        sA[nf][r] = pv;
        rs += pv;
      }
#pragma unroll
      for (int off = 1; off < 16; off <<= 1) rs += __shfl_xor(rs, off, 32);
      li[r] = li[r] * al + rs;
#pragma unroll
      for (int nn = 0; nn < 4; nn++) o[nn][r] *= al;
    }

    // P -> LDS (per-wave region), then O += P * V (batch-preloaded frags).
#pragma unroll
    for (int nf = 0; nf < 4; nf++)
#pragma unroll
      for (int r = 0; r < 8; r++)
        Ps[(wv * 16 + r + (lhi << 3)) * LD + nf * 16 + lrow] = (_Float16)sA[nf][r];

#pragma unroll
    for (int ks = 0; ks < 2; ks++) {
      v16h ap = frag16(&Ps[(wv * 16 + lrow) * LD + ks * 32 + lhi * 8], 16);
      v16h bv[4];
#pragma unroll
      for (int nn = 0; nn < 4; nn++)
        bv[nn] = frag16(&Vs[(nn * 16 + lrow) * LD + ks * 32 + lhi * 8], 16);
#pragma unroll
      for (int nn = 0; nn < 4; nn++)
        o[nn] = __builtin_amdgcn_wmma_f32_16x16x32_f16(
            false, ap, false, bv[nn], (short)0, o[nn], false, false);
    }
  }

  // Store ctx with torch's stack(dim=-1) interleave: col = d*H + h
#pragma unroll
  for (int nn = 0; nn < 4; nn++)
#pragma unroll
    for (int r = 0; r < 8; r++) {
      int sl = qb * 64 + wv * 16 + r + (lhi << 3);
      if (sl < S_) {
        int d = nn * 16 + lrow;
        float v = o[nn][r] / li[r];
        ctx[((size_t)(bb * S_ + sl)) * DV_ + d * H_ + hh] = (_Float16)v;
      }
    }
}

// ---------------------------------------------------------------------------
// Host launcher
// ---------------------------------------------------------------------------
extern "C" void kernel_launch(void* const* d_in, const int* in_sizes, int n_in,
                              void* d_out, int out_size, void* d_ws, size_t ws_size,
                              hipStream_t stream) {
  (void)in_sizes; (void)n_in; (void)out_size; (void)ws_size;
  const float* x     = (const float*)d_in[0];
  const float* encW  = (const float*)d_in[1];
  const float* encB  = (const float*)d_in[2];
  const float* Wq    = (const float*)d_in[3];
  const float* bq    = (const float*)d_in[4];
  const float* Wk    = (const float*)d_in[5];
  const float* bk    = (const float*)d_in[6];
  const float* Wv    = (const float*)d_in[7];
  const float* bv    = (const float*)d_in[8];
  const float* Wo    = (const float*)d_in[9];
  const float* bo    = (const float*)d_in[10];
  const float* W1    = (const float*)d_in[11];
  const float* b1    = (const float*)d_in[12];
  const float* W2    = (const float*)d_in[13];
  const float* b2    = (const float*)d_in[14];
  const float* ln1w  = (const float*)d_in[15];
  const float* ln1b  = (const float*)d_in[16];
  const float* ln2w  = (const float*)d_in[17];
  const float* ln2b  = (const float*)d_in[18];
  const float* outW  = (const float*)d_in[19];
  const float* outB  = (const float*)d_in[20];
  float* out = (float*)d_out;

  // Workspace layout
  const size_t NE = (size_t)BS_ * DV_;            // 19,660,800 elems
  char* w = (char*)d_ws;
  float*    h32 = (float*)w;                      // [BS,DV] f32 hidden
  float*    t32 = (float*)(w + NE * 4);           // [BS,DV] f32 tmp (GEMM out)
  _Float16* a16 = (_Float16*)(w + NE * 8);        // f16 activation copy
  _Float16* q16 = a16 + NE;                       // [B,H,S,DH] f16
  _Float16* k16 = q16 + NE;
  _Float16* v16 = k16 + NE;
  _Float16* c16 = v16 + NE;                       // ctx / ELU intermediate
  _Float16* wt  = c16 + NE;                       // one transposed f16 weight (2 MB)

  const int nElem   = (int)NE;                    // fits in int
  const int gElem   = nElem / 256;                // 76800
  const int gWeight = (DV_ * DV_) / 256;          // 4096
  dim3 gGemm(DV_ / 128, BS_ / 128);               // (8, 150)
  dim3 gAttn((S_ + 63) / 64, H_, B_);             // (10, 16, 32)

  k_enc_in<<<gElem, 256, 0, stream>>>(x, encW, encB, h32);

  for (int l = 0; l < L_; l++) {
    const size_t wstep = (size_t)DV_ * DV_;       // per-layer weight stride
    k_f32_to_f16<<<gElem, 256, 0, stream>>>(h32, a16, nElem);

    k_qkv_w_to_f16_T<<<gWeight, 256, 0, stream>>>(Wq + l * wstep, wt);
    k_gemm<1><<<gGemm, 256, 0, stream>>>(a16, wt, bq + l * DV_, q16, BS_, DV_, DV_);
    k_qkv_w_to_f16_T<<<gWeight, 256, 0, stream>>>(Wk + l * wstep, wt);
    k_gemm<1><<<gGemm, 256, 0, stream>>>(a16, wt, bk + l * DV_, k16, BS_, DV_, DV_);
    k_qkv_w_to_f16_T<<<gWeight, 256, 0, stream>>>(Wv + l * wstep, wt);
    k_gemm<1><<<gGemm, 256, 0, stream>>>(a16, wt, bv + l * DV_, v16, BS_, DV_, DV_);

    k_attn<<<gAttn, 128, 0, stream>>>(q16, k16, v16, c16);

    k_w_to_f16_T<<<gWeight, 256, 0, stream>>>(Wo + l * wstep, wt);
    k_gemm<0><<<gGemm, 256, 0, stream>>>(c16, wt, bo + l * DV_, t32, BS_, DV_, DV_);

    k_add_ln<<<BS_, 256, 0, stream>>>(h32, t32, ln1w + l * DV_, ln1b + l * DV_, h32, a16);

    // FFN: elu(x1 @ W2 + b2) @ W1 + b1
    k_w_to_f16_T<<<gWeight, 256, 0, stream>>>(W2 + l * wstep, wt);
    k_gemm<2><<<gGemm, 256, 0, stream>>>(a16, wt, b2 + l * DV_, c16, BS_, DV_, DV_);
    k_w_to_f16_T<<<gWeight, 256, 0, stream>>>(W1 + l * wstep, wt);
    k_gemm<0><<<gGemm, 256, 0, stream>>>(c16, wt, b1 + l * DV_, t32, BS_, DV_, DV_);

    k_add_ln<<<BS_, 256, 0, stream>>>(h32, t32, ln2w + l * DV_, ln2b + l * DV_, h32, a16);
  }

  k_out_proj<<<(BS_ * F_ + 255) / 256, 256, 0, stream>>>(h32, outW, outB, out);
}
